// KeyPointRelationNet_3410204033181
// MI455X (gfx1250) — matrix operand order repaired and checked by reading, not compiled, not run on hardware
//
#include <hip/hip_runtime.h>
#include <stdint.h>

// ---------------- problem constants ----------------
#define BB   8
#define CC   256
#define HH   128
#define WW   128
#define NKP  512
#define NEDG 4096
#define NN   (BB*NKP)            // 4096 nodes
#define BE   (BB*NEDG)           // 32768 batched edges
#define ETOT (BE + NN)           // + self loops = 36864
#define HEADS 3
#define KRAW 288                 // 258 padded up to a multiple of 32

// ---------------- WMMA types ----------------
typedef __attribute__((ext_vector_type(16))) __bf16 v16bf;
typedef __attribute__((ext_vector_type(8)))  float  v8f;

union Frag { unsigned int u[8]; v16bf v; };

__device__ __forceinline__ unsigned short f2bf(float f) {
  union { float f; unsigned int u; } c; c.f = f;
  unsigned int u = c.u;
  u += 0x7FFFu + ((u >> 16) & 1u);          // round-to-nearest-even
  return (unsigned short)(u >> 16);
}
__device__ __forceinline__ unsigned int pack2(float x, float y) {
  return (unsigned int)f2bf(x) | ((unsigned int)f2bf(y) << 16);
}

// ---------------- weight pre-pack: f32 [N x K] -> bf16 [N x Kp], zero-padded ----------------
__global__ void pack_bf16(const float* __restrict__ src, unsigned short* __restrict__ dst,
                          int N, int K, int Kp)
{
  long i = (long)blockIdx.x * blockDim.x + threadIdx.x;
  long total = (long)N * Kp;
  if (i >= total) return;
  int n = (int)(i / Kp), k = (int)(i % Kp);
  float v = (k < K) ? src[(long)n * K + k] : 0.f;
  dst[i] = f2bf(v);
}

// ---------------- GEMM: C = act(A[MxK] * Wb[NxK]^T + bias) ----------------
// Requirements (guaranteed by caller): M%64==0, N%64==0, K%32==0, lda%32==0.
// 256 threads = 8 waves; block tile 64x64; each wave: 1 A-frag, 2 B-frags, 2 WMMAs/k-step.
__global__ __launch_bounds__(256) void gemm_wmma(
    const float* __restrict__ A, int lda,
    const unsigned short* __restrict__ Wb,    // bf16 packed, [N x K]
    const float* __restrict__ bias,
    float* __restrict__ C, int ldc,
    int M, int N, int K, float slope)         // slope=1 -> linear
{
  __shared__ unsigned short lA[64][40];       // 80B rows: 16B-aligned, bank-spread
  __shared__ unsigned short lB[64][40];

  const int tid  = threadIdx.x;
  const int lane = tid & 31;
  const int wave = tid >> 5;
  const int wm   = wave >> 1;                 // 0..3  (16-row M subtile)
  const int wn   = wave & 1;                  // 0..1  (32-col N half, 2x16 subtiles)
  const int m16  = lane & 15;
  const int half = lane >> 4;
  const long bM  = (long)blockIdx.x * 64;
  const long bN  = (long)blockIdx.y * 64;

  // staging: thread t -> row t/4 (0..63), col group (t%4)*8
  const int sr = tid >> 2;
  const int sc = (tid & 3) * 8;
  const float* __restrict__          Aptr = A  + (bM + sr) * lda + sc;
  const unsigned short* __restrict__ Wptr = Wb + (bN + sr) * (long)K + sc;

  v8f acc0 = {}, acc1 = {};

  for (int k0 = 0; k0 < K; k0 += 32) {
    // stage A tile (64x32 f32 -> bf16): two b128 loads, 4 packed u32 LDS stores
    float4 a0 = *(const float4*)(Aptr + k0);
    float4 a1 = *(const float4*)(Aptr + k0 + 4);
    unsigned int* pa = (unsigned int*)&lA[sr][sc];
    pa[0] = pack2(a0.x, a0.y);
    pa[1] = pack2(a0.z, a0.w);
    pa[2] = pack2(a1.x, a1.y);
    pa[3] = pack2(a1.z, a1.w);
    // stage B tile (64x32 bf16): one b128 copy per thread
    *(uint4*)&lB[sr][sc] = *(const uint4*)(Wptr + k0);
    // prefetch next k-step tiles while WMMAs run
    if (k0 + 32 < K) {
      __builtin_prefetch(Aptr + k0 + 32, 0, 0);
      __builtin_prefetch(Wptr + k0 + 32, 0, 0);
    }
    __syncthreads();

    // A fragment: lane m16, VGPR i -> K = 16*(i>>2) + 8*half + 2*(i&3)
    Frag af, bf0, bf1;
    const int arow = wm * 16 + m16;
#pragma unroll
    for (int i = 0; i < 8; ++i) {
      int kk = ((i >> 2) << 4) + (half << 3) + ((i & 3) << 1);
      af.u[i] = *(const unsigned int*)&lA[arow][kk];
    }
    // B fragments: lane n=m16, VGPR i -> K = 16*(i>>2) + 8*half + 2*(i&3) ; two N subtiles
    const int brow = wn * 32 + m16;
#pragma unroll
    for (int i = 0; i < 8; ++i) {
      int kk = ((i >> 2) << 4) + (half << 3) + ((i & 3) << 1);
      bf0.u[i] = *(const unsigned int*)&lB[brow][kk];
      bf1.u[i] = *(const unsigned int*)&lB[brow + 16][kk];
    }

    acc0 = __builtin_amdgcn_wmma_f32_16x16x32_bf16(false, af.v, false, bf0.v,
                                                   (short)0, acc0, false, false);
    acc1 = __builtin_amdgcn_wmma_f32_16x16x32_bf16(false, af.v, false, bf1.v,
                                                   (short)0, acc1, false, false);
    __syncthreads();
  }

  // C/D layout: VGPR i -> M = 8*half + i ; N = lane&15
  const long gn0 = bN + wn * 32 + m16;
  const long gn1 = gn0 + 16;
  const float bv0 = bias ? bias[gn0] : 0.f;
  const float bv1 = bias ? bias[gn1] : 0.f;
#pragma unroll
  for (int i = 0; i < 8; ++i) {
    long gm = bM + wm * 16 + (half << 3) + i;
    float v0 = acc0[i] + bv0;
    float v1 = acc1[i] + bv1;
    v0 = (v0 > 0.f) ? v0 : v0 * slope;
    v1 = (v1 > 0.f) ? v1 : v1 * slope;
    C[gm * ldc + gn0] = v0;
    C[gm * ldc + gn1] = v1;
  }
}

// ---------------- gather conv inputs only at keypoints ----------------
__global__ void gather_kernel(const float* __restrict__ hm1, const float* __restrict__ hm2,
                              const float* __restrict__ hm3, const int* __restrict__ coords,
                              float* __restrict__ Xg)
{
  int node = blockIdx.x;                 // 0..NN-1
  int b = node >> 9;                     // /NKP
  int y = coords[node * 2 + 0];
  int x = coords[node * 2 + 1];
  long base = ((long)b * CC) * (HH * WW) + (long)y * WW + x;
  float* out = Xg + (long)node * 768;
  for (int c = threadIdx.x; c < CC; c += blockDim.x) {
    long off = base + (long)c * (HH * WW);
    out[c]       = hm1[off];
    out[256 + c] = hm2[off];
    out[512 + c] = hm3[off];
  }
}

__global__ void append_norms(const float* __restrict__ norms, float* __restrict__ raw)
{
  int node = blockIdx.x * blockDim.x + threadIdx.x;
  if (node < NN) {
    float* r = raw + (long)node * KRAW;
    r[256] = norms[node * 2 + 0];
    r[257] = norms[node * 2 + 1];
#pragma unroll
    for (int k = 258; k < KRAW; ++k) r[k] = 0.f;   // K padding must be zero
  }
}

// ---------------- batched edge list with self loops ----------------
__global__ void build_edges(const int* __restrict__ edges, int* __restrict__ src, int* __restrict__ dst)
{
  int i = blockIdx.x * blockDim.x + threadIdx.x;
  if (i < BE) {
    int b = i / NEDG;
    src[i] = edges[(long)i * 2 + 0] + b * NKP;
    dst[i] = edges[(long)i * 2 + 1] + b * NKP;
  } else if (i < ETOT) {
    int s = i - BE;
    src[i] = s; dst[i] = s;
  }
}

// ---------------- attention: per-node el/er (one wave per node*head) ----------------
__global__ void attn_scores(const float* __restrict__ h, const float* __restrict__ al,
                            const float* __restrict__ ar, float* __restrict__ el,
                            float* __restrict__ er, int outD)
{
  int wid  = (blockIdx.x * blockDim.x + threadIdx.x) >> 5;
  int lane = threadIdx.x & 31;
  if (wid >= NN * HEADS) return;
  int node = wid / HEADS, hd = wid % HEADS;
  const float* hp  = h + (long)node * HEADS * outD + hd * outD;
  const float* alp = al + hd * outD;
  const float* arp = ar + hd * outD;
  float sl = 0.f, sr = 0.f;
  for (int d = lane; d < outD; d += 32) { float v = hp[d]; sl += v * alp[d]; sr += v * arp[d]; }
#pragma unroll
  for (int off = 16; off > 0; off >>= 1) {
    sl += __shfl_down(sl, off, 32);
    sr += __shfl_down(sr, off, 32);
  }
  if (lane == 0) { el[wid] = sl; er[wid] = sr; }
}

__global__ void init_attn(unsigned int* __restrict__ maxu, float* __restrict__ sum,
                          float* __restrict__ agg, long aggN)
{
  long i = (long)blockIdx.x * blockDim.x + threadIdx.x;
  if (i < NN * HEADS) { maxu[i] = 0u; sum[i] = 0.f; }
  if (i < aggN) agg[i] = 0.f;
}

// monotonic float<->uint order mapping for atomicMax
__device__ __forceinline__ unsigned int f2ord(float f) {
  union { float f; unsigned int u; } c; c.f = f;
  return c.u ^ ((((int)c.u) >> 31) | 0x80000000u);
}
__device__ __forceinline__ float ord2f(unsigned int u) {
  unsigned int b = (u & 0x80000000u) ? (u ^ 0x80000000u) : ~u;
  union { unsigned int u; float f; } c; c.u = b;
  return c.f;
}

__global__ void edge_max(const int* __restrict__ src, const int* __restrict__ dst,
                         const float* __restrict__ el, const float* __restrict__ er,
                         unsigned int* __restrict__ maxu)
{
  int i = blockIdx.x * blockDim.x + threadIdx.x;
  if (i >= ETOT * HEADS) return;
  int e = i / HEADS, hd = i % HEADS;
  float v = el[src[e] * HEADS + hd] + er[dst[e] * HEADS + hd];
  v = (v > 0.f) ? v : 0.2f * v;                       // leaky(0.2)
  atomicMax(&maxu[dst[e] * HEADS + hd], f2ord(v));
}

__global__ void edge_expsum(const int* __restrict__ src, const int* __restrict__ dst,
                            const float* __restrict__ el, const float* __restrict__ er,
                            const unsigned int* __restrict__ maxu, float* __restrict__ ex,
                            float* __restrict__ sum)
{
  int i = blockIdx.x * blockDim.x + threadIdx.x;
  if (i >= ETOT * HEADS) return;
  int e = i / HEADS, hd = i % HEADS;
  int d = dst[e];
  float v = el[src[e] * HEADS + hd] + er[d * HEADS + hd];
  v = (v > 0.f) ? v : 0.2f * v;
  float x = __expf(v - ord2f(maxu[d * HEADS + hd]));
  ex[i] = x;
  atomicAdd(&sum[d * HEADS + hd], x);
}

__global__ void edge_agg(const int* __restrict__ src, const int* __restrict__ dst,
                         const float* __restrict__ ex, const float* __restrict__ sum,
                         const float* __restrict__ h, float* __restrict__ agg, int outD)
{
  int e = blockIdx.x;
  int d = dst[e], s = src[e];
  int tot = HEADS * outD;
  for (int t = threadIdx.x; t < tot; t += blockDim.x) {
    int hd = t / outD;
    float alpha = ex[e * HEADS + hd] / sum[d * HEADS + hd];
    atomicAdd(&agg[(long)d * tot + t], h[(long)s * tot + t] * alpha);
  }
}

__global__ void add_bias(float* __restrict__ agg, const float* __restrict__ bias, int F, long total)
{
  long i = (long)blockIdx.x * blockDim.x + threadIdx.x;
  if (i < total) agg[i] += bias[i % F];
}

// ---------------- per-edge pair classification ----------------
__global__ void edge_cls(const float* __restrict__ ft, const int* __restrict__ edges,
                         const float* __restrict__ ew, const float* __restrict__ eb,
                         float* __restrict__ out)
{
  int i = blockIdx.x * blockDim.x + threadIdx.x;     // one per (b,e)
  if (i >= BE) return;
  int b = i >> 12;                                   // /NEDG
  const float* fi = ft + (long)(edges[(long)i * 2 + 0] + b * NKP) * 128;
  const float* fj = ft + (long)(edges[(long)i * 2 + 1] + b * NKP) * 128;
  float a0 = eb[0], a1 = eb[1];
  for (int c = 0; c < 128; c += 4) {
    float4 vi = *(const float4*)(fi + c);
    float4 vj = *(const float4*)(fj + c);
    float4 w0i = *(const float4*)(ew + c);
    float4 w0j = *(const float4*)(ew + 128 + c);
    float4 w1i = *(const float4*)(ew + 256 + c);
    float4 w1j = *(const float4*)(ew + 384 + c);
    a0 += vi.x*w0i.x + vi.y*w0i.y + vi.z*w0i.z + vi.w*w0i.w
        + vj.x*w0j.x + vj.y*w0j.y + vj.z*w0j.z + vj.w*w0j.w;
    a1 += vi.x*w1i.x + vi.y*w1i.y + vi.z*w1i.z + vi.w*w1i.w
        + vj.x*w1j.x + vj.y*w1j.y + vj.z*w1j.z + vj.w*w1j.w;
  }
  out[(long)i * 2 + 0] = a0;
  out[(long)i * 2 + 1] = a1;
}

// ---------------- host side ----------------
static inline void launch_gemm(const float* A, int lda, const unsigned short* Wb,
                               const float* bias, float* C, int ldc,
                               int M, int N, int K, float slope, hipStream_t s)
{
  dim3 grid(M / 64, N / 64);
  gemm_wmma<<<grid, dim3(256), 0, s>>>(A, lda, Wb, bias, C, ldc, M, N, K, slope);
}

static inline void launch_pack(const float* src, unsigned short* dst, int N, int K, int Kp,
                               hipStream_t s)
{
  long total = (long)N * Kp;
  pack_bf16<<<(unsigned)((total + 255) / 256), 256, 0, s>>>(src, dst, N, K, Kp);
}

struct LayerScratch {
  float *h, *agg, *el, *er, *sumb, *ex;
  unsigned int* maxu;
  int *src, *dst;
};

static void run_gat_layer(const float* ftin, int inF, int outD,
                          const unsigned short* gWp, const float* gal, const float* gar,
                          const float* gb, const unsigned short* gateWp, const float* gateB,
                          float* ftout, int gateOut,
                          const LayerScratch& S, hipStream_t s)
{
  const int outTot = HEADS * outD;
  const long aggN = (long)NN * outTot;
  // h = ft @ gW^T
  launch_gemm(ftin, inF, gWp, nullptr, S.h, outTot, NN, outTot, inF, 1.0f, s);
  attn_scores<<<(NN * HEADS * 32 + 255) / 256, 256, 0, s>>>(S.h, gal, gar, S.el, S.er, outD);
  {
    long tot = aggN > (NN * HEADS) ? aggN : (NN * HEADS);
    init_attn<<<(unsigned)((tot + 255) / 256), 256, 0, s>>>(S.maxu, S.sumb, S.agg, aggN);
  }
  edge_max<<<(ETOT * HEADS + 255) / 256, 256, 0, s>>>(S.src, S.dst, S.el, S.er, S.maxu);
  edge_expsum<<<(ETOT * HEADS + 255) / 256, 256, 0, s>>>(S.src, S.dst, S.el, S.er, S.maxu, S.ex, S.sumb);
  edge_agg<<<ETOT, 256, 0, s>>>(S.src, S.dst, S.ex, S.sumb, S.h, S.agg, outD);
  add_bias<<<(unsigned)((aggN + 255) / 256), 256, 0, s>>>(S.agg, gb, outTot, aggN);
  // ftout = leaky(agg @ gateW^T + gateB, 0.01)
  launch_gemm(S.agg, outTot, gateWp, gateB, ftout, gateOut, NN, gateOut, outTot, 0.01f, s);
}

extern "C" void kernel_launch(void* const* d_in, const int* in_sizes, int n_in,
                              void* d_out, int out_size, void* d_ws, size_t ws_size,
                              hipStream_t stream)
{
  (void)in_sizes; (void)n_in; (void)out_size; (void)ws_size;
  const float* hm1    = (const float*)d_in[0];
  const float* hm2    = (const float*)d_in[1];
  const float* hm3    = (const float*)d_in[2];
  const int*   coords = (const int*)  d_in[3];
  const float* norms  = (const float*)d_in[4];
  const int*   edges  = (const int*)  d_in[5];
  const float* m4_w   = (const float*)d_in[6];
  const float* m4_b   = (const float*)d_in[7];
  const float* emb_w  = (const float*)d_in[8];
  const float* emb_b  = (const float*)d_in[9];
  const float* g1_w   = (const float*)d_in[10];
  const float* g1_al  = (const float*)d_in[11];
  const float* g1_ar  = (const float*)d_in[12];
  const float* g1_b   = (const float*)d_in[13];
  const float* gate1_w= (const float*)d_in[14];
  const float* gate1_b= (const float*)d_in[15];
  const float* g2_w   = (const float*)d_in[16];
  const float* g2_al  = (const float*)d_in[17];
  const float* g2_ar  = (const float*)d_in[18];
  const float* g2_b   = (const float*)d_in[19];
  const float* gate2_w= (const float*)d_in[20];
  const float* gate2_b= (const float*)d_in[21];
  const float* g3_w   = (const float*)d_in[22];
  const float* g3_al  = (const float*)d_in[23];
  const float* g3_ar  = (const float*)d_in[24];
  const float* g3_b   = (const float*)d_in[25];
  const float* gate3_w= (const float*)d_in[26];
  const float* gate3_b= (const float*)d_in[27];
  const float* edge_w = (const float*)d_in[28];
  const float* edge_b = (const float*)d_in[29];
  float* out = (float*)d_out;

  // workspace carve-up (deterministic every call)
  char* w = (char*)d_ws;
  auto alloc = [&](size_t bytes) -> char* {
    char* p = w; w += (bytes + 255) & ~(size_t)255; return p;
  };
  float* Xg   = (float*)alloc((size_t)NN * 768 * 4);   // reused as agg after consumption
  float* hbuf = (float*)alloc((size_t)NN * 768 * 4);
  float* raw  = (float*)alloc((size_t)NN * KRAW * 4);
  float* node = (float*)alloc((size_t)NN * 128 * 4);
  float* ftA  = (float*)alloc((size_t)NN * 256 * 4);
  float* ftB  = (float*)alloc((size_t)NN * 256 * 4);
  float* el   = (float*)alloc((size_t)NN * HEADS * 4);
  float* er   = (float*)alloc((size_t)NN * HEADS * 4);
  unsigned int* maxu = (unsigned int*)alloc((size_t)NN * HEADS * 4);
  float* sumb = (float*)alloc((size_t)NN * HEADS * 4);
  float* ex   = (float*)alloc((size_t)ETOT * HEADS * 4);
  int*   srcv = (int*)alloc((size_t)ETOT * 4);
  int*   dstv = (int*)alloc((size_t)ETOT * 4);
  // bf16-packed weights
  unsigned short* m4p    = (unsigned short*)alloc((size_t)256 * 768 * 2);
  unsigned short* embp   = (unsigned short*)alloc((size_t)128 * KRAW * 2);
  unsigned short* g1p    = (unsigned short*)alloc((size_t)768 * 128 * 2);
  unsigned short* gate1p = (unsigned short*)alloc((size_t)256 * 768 * 2);
  unsigned short* g2p    = (unsigned short*)alloc((size_t)768 * 256 * 2);
  unsigned short* gate2p = (unsigned short*)alloc((size_t)256 * 768 * 2);
  unsigned short* g3p    = (unsigned short*)alloc((size_t)384 * 256 * 2);
  unsigned short* gate3p = (unsigned short*)alloc((size_t)128 * 384 * 2);

  // 0) pack weights to bf16 (K padded where needed)
  launch_pack(m4_w,    m4p,    256, 768, 768,  stream);
  launch_pack(emb_w,   embp,   128, 258, KRAW, stream);
  launch_pack(g1_w,    g1p,    768, 128, 128,  stream);
  launch_pack(gate1_w, gate1p, 256, 768, 768,  stream);
  launch_pack(g2_w,    g2p,    768, 256, 256,  stream);
  launch_pack(gate2_w, gate2p, 256, 768, 768,  stream);
  launch_pack(g3_w,    g3p,    384, 256, 256,  stream);
  launch_pack(gate3_w, gate3p, 128, 384, 384,  stream);

  // 1) gather 768-chan features only at the 4096 keypoints (skip full conv!)
  gather_kernel<<<NN, 256, 0, stream>>>(hm1, hm2, hm3, coords, Xg);
  // 2) fp = Xg @ m4_w^T + m4_b  (written into raw with row stride KRAW)
  launch_gemm(Xg, 768, m4p, m4_b, raw, KRAW, NN, 256, 768, 1.0f, stream);
  // 3) append norms + zero K padding -> raw [NN, KRAW]
  append_norms<<<(NN + 255) / 256, 256, 0, stream>>>(norms, raw);
  // 4) node = raw @ emb_w^T + emb_b
  launch_gemm(raw, KRAW, embp, emb_b, node, 128, NN, 128, KRAW, 1.0f, stream);
  // 5) batched edge list with self loops
  build_edges<<<(ETOT + 255) / 256, 256, 0, stream>>>(edges, srcv, dstv);

  LayerScratch S{hbuf, /*agg=*/Xg, el, er, sumb, ex, maxu, srcv, dstv};
  // GAT1: 128 -> 3x256, gate -> 256
  run_gat_layer(node, 128, 256, g1p, g1_al, g1_ar, g1_b, gate1p, gate1_b, ftA, 256, S, stream);
  // GAT2: 256 -> 3x256, gate -> 256
  run_gat_layer(ftA, 256, 256, g2p, g2_al, g2_ar, g2_b, gate2p, gate2_b, ftB, 256, S, stream);
  // GAT3: 256 -> 3x128, gate -> 128
  run_gat_layer(ftB, 256, 128, g3p, g3_al, g3_ar, g3_b, gate3p, gate3_b, ftA, 128, S, stream);

  // edge classification
  edge_cls<<<(BE + 255) / 256, 256, 0, stream>>>(ftA, edges, edge_w, edge_b, out);
}
